// MultiHeadConvNNAttention_39977555591746
// MI455X (gfx1250) — matrix operand order repaired and checked by reading, not compiled, not run on hardware
//
#include <hip/hip_runtime.h>
#include <hip/hip_bf16.h>

// ---------------------------------------------------------------------------
// Problem constants (from reference): B=4, S=2048, D=512, H=8, K=9, DK=64
// ---------------------------------------------------------------------------
#define BB   4
#define SS   2048
#define DD   512
#define HH   8
#define KK_TOP 9
#define DK   64
#define BH   32          // B*H
#define MM   8192        // B*S
#define GEMM_K 512

typedef __attribute__((ext_vector_type(16))) __bf16 v16bf;
typedef __attribute__((ext_vector_type(8)))  __bf16 v8bf;
typedef __attribute__((ext_vector_type(8)))  float  v8f;
typedef int vi4 __attribute__((vector_size(16)));   // matches builtin param type

#define AS1 __attribute__((address_space(1)))
#define AS3 __attribute__((address_space(3)))

// CDNA5 async memory->LDS path (ASYNCcnt-tracked, bypasses VGPRs)
#if defined(__AMDGCN__) && \
    __has_builtin(__builtin_amdgcn_global_load_async_to_lds_b128) && \
    __has_builtin(__builtin_amdgcn_global_load_async_to_lds_b32) && \
    __has_builtin(__builtin_amdgcn_s_wait_asynccnt)
#define USE_ASYNC_LDS 1
#else
#define USE_ASYNC_LDS 0
#endif

__device__ inline void wait_async_all() {
#if USE_ASYNC_LDS
    __builtin_amdgcn_s_wait_asynccnt(0);
#endif
}

// fp32 -> bf16 round-to-nearest-even
__device__ inline __bf16 f2bf(float f) {
    union { float f; unsigned u; } v; v.f = f;
    unsigned r = v.u + 0x7FFFu + ((v.u >> 16) & 1u);
    unsigned short h = (unsigned short)(r >> 16);
    return __builtin_bit_cast(__bf16, h);
}

__device__ inline v16bf make_frag(const __bf16* lo, const __bf16* hi) {
    v8bf l = *(const v8bf*)lo;
    v8bf h = *(const v8bf*)hi;
    v16bf r;
#pragma unroll
    for (int i = 0; i < 8; ++i) { r[i] = l[i]; r[i + 8] = h[i]; }
    return r;
}

__device__ inline v8f wmma_bf16(v16bf a, v16bf b, v8f c) {
    // D = A(16x32 bf16) * B(32x16 bf16) + C(16x16 f32)
    return __builtin_amdgcn_wmma_f32_16x16x32_bf16(
        /*neg_a=*/false, a, /*neg_b=*/false, b,
        /*c_mod=*/(short)0, c, /*reuse_a=*/false, /*reuse_b=*/false);
}

// A-fragment (16x32 bf16, M x K) from LDS row-major [row][stride] at k-slice kt*32
// lane L: M = L&15 ; K = kt*32 + ((L<16)?0:8) + e (+8 for e>=8)  -> two 16B runs
template <int STRIDE>
__device__ inline v16bf frag_a(const __bf16* lds, int row0, int kt, int lane) {
    int m  = row0 + (lane & 15);
    int kb = kt * 32 + ((lane < 16) ? 0 : 8);
    const __bf16* p = lds + m * STRIDE + kb;
    return make_frag(p, p + 16);
}

// B-fragment (32x16 bf16, K x N) from LDS stored as [n][k] row-major
// lane L: N = L&15 ; K = kt*32 + ((L<16)?0:8) + e (+8 for e>=8)  -> two 16B runs
template <int STRIDE>
__device__ inline v16bf frag_b(const __bf16* lds, int col0, int kt, int lane) {
    int n  = col0 + (lane & 15);
    int ko = kt * 32 + ((lane < 16) ? 0 : 8);
    const __bf16* p = lds + n * STRIDE + ko;
    return make_frag(p, p + 16);
}

// ---------------------------------------------------------------------------
// 1) fp32 -> bf16 conversion
// ---------------------------------------------------------------------------
__global__ void cvt_bf16_kernel(const float* __restrict__ src,
                                __bf16* __restrict__ dst, int n) {
    int t = blockIdx.x * blockDim.x + threadIdx.x;
    if (t < n) dst[t] = f2bf(src[t]);
}

// ---------------------------------------------------------------------------
// 2) GEMM: C[m,n] = sum_k A[m,k] * W[n,k] + bias[n]
//    A: bf16 [MM][512] row-major, W: bf16 [512][512] row-major (n-major)
//    mode 0: dst fp32 [MM][512] row-major
//    mode 1: dst fp32 in head layout [BH][DK][SS] ( _to_bh baked in )
//    WG = 128 threads (4 waves), 64x64 tile, each wave 32x32 via 2x2 WMMA.
//    Double-buffered LDS; k-slice t+1 streamed by the async engine while
//    slice t is consumed by WMMA.
// ---------------------------------------------------------------------------
#define LDSA 40   // lds row stride in bf16 (80B: 16B aligned, conflict-spread)

__device__ inline void stage_slice(const __bf16* __restrict__ A,
                                   const __bf16* __restrict__ W,
                                   __bf16* sA, __bf16* sB,
                                   int m0, int n0, int kt, int tid) {
#pragma unroll
    for (int it = 0; it < 2; ++it) {
        int t  = tid + it * 128;
        int r  = t >> 2;
        int cc = (t & 3) * 8;
        const __bf16* ga = &A[(size_t)(m0 + r) * GEMM_K + kt + cc];
        const __bf16* gw = &W[(size_t)(n0 + r) * GEMM_K + kt + cc];
#if USE_ASYNC_LDS
        __builtin_amdgcn_global_load_async_to_lds_b128(
            (AS1 vi4*)ga, (AS3 vi4*)&sA[r * LDSA + cc], 0, 0);
        __builtin_amdgcn_global_load_async_to_lds_b128(
            (AS1 vi4*)gw, (AS3 vi4*)&sB[r * LDSA + cc], 0, 0);
#else
        *(v8bf*)&sA[r * LDSA + cc] = *(const v8bf*)ga;
        *(v8bf*)&sB[r * LDSA + cc] = *(const v8bf*)gw;
#endif
    }
}

__global__ __launch_bounds__(128) void gemm_bf16_kernel(
    const __bf16* __restrict__ A, const __bf16* __restrict__ W,
    const float* __restrict__ bias, float* __restrict__ dst, int mode) {
    __shared__ __align__(16) __bf16 As[2][64 * LDSA];
    __shared__ __align__(16) __bf16 Bs[2][64 * LDSA];

    const int tid  = threadIdx.x;
    const int lane = tid & 31;
    const int wave = tid >> 5;
    const int n0 = blockIdx.x * 64;
    const int m0 = blockIdx.y * 64;
    const int wr = (wave >> 1) * 32;   // wave row within tile
    const int wc = (wave & 1) * 32;    // wave col within tile

    v8f acc[2][2] = {};

    // prologue: stage k-slice 0
    stage_slice(A, W, As[0], Bs[0], m0, n0, 0, tid);
    wait_async_all();
    __syncthreads();

    int buf = 0;
    for (int kt = 0; kt < GEMM_K; kt += 32) {
        // stream next slice through the async engine while we compute
        if (kt + 32 < GEMM_K)
            stage_slice(A, W, As[buf ^ 1], Bs[buf ^ 1], m0, n0, kt + 32, tid);
        if (kt + 64 < GEMM_K)  // pull the slice after that toward L2
            __builtin_prefetch(&A[(size_t)(m0 + (tid >> 1)) * GEMM_K + kt + 64], 0, 0);

        v16bf a0 = frag_a<LDSA>(As[buf], wr,      0, lane);
        v16bf a1 = frag_a<LDSA>(As[buf], wr + 16, 0, lane);
        v16bf b0 = frag_b<LDSA>(Bs[buf], wc,      0, lane);
        v16bf b1 = frag_b<LDSA>(Bs[buf], wc + 16, 0, lane);

        acc[0][0] = wmma_bf16(a0, b0, acc[0][0]);
        acc[0][1] = wmma_bf16(a0, b1, acc[0][1]);
        acc[1][0] = wmma_bf16(a1, b0, acc[1][0]);
        acc[1][1] = wmma_bf16(a1, b1, acc[1][1]);

        wait_async_all();
        __syncthreads();
        buf ^= 1;
    }

    // epilogue: C layout per ISA — VGPR r: lanes 0-15 (M=r), lanes 16-31 (M=8+r), N=lane&15
#pragma unroll
    for (int sm = 0; sm < 2; ++sm) {
#pragma unroll
        for (int sn = 0; sn < 2; ++sn) {
            int n  = n0 + wc + sn * 16 + (lane & 15);
            int mb = m0 + wr + sm * 16 + ((lane < 16) ? 0 : 8);
            float bv = bias[n];
#pragma unroll
            for (int r = 0; r < 8; ++r) {
                int   m   = mb + r;
                float val = acc[sm][sn][r] + bv;
                if (mode == 0) {
                    dst[(size_t)m * DD + n] = val;
                } else {
                    int b = m >> 11, s = m & (SS - 1);     // m = b*S + s
                    int h = n >> 6,  c = n & (DK - 1);     // n = h*DK + c
                    dst[(size_t)(((b << 3) + h) * DK + c) * SS + s] = val;
                }
            }
        }
    }
}

// ---------------------------------------------------------------------------
// 3) channel-normalize [BH][DK][SS] fp32 -> bf16 (norm over DK, eps clamp)
// ---------------------------------------------------------------------------
__global__ __launch_bounds__(256) void normalize_kernel(
    const float* __restrict__ src, __bf16* __restrict__ dst) {
    int s  = blockIdx.x * blockDim.x + threadIdx.x;   // 0..SS-1
    int bh = blockIdx.y;
    const float* p = src + (size_t)bh * DK * SS + s;
    float sum = 0.f;
#pragma unroll 8
    for (int c = 0; c < DK; ++c) { float v = p[(size_t)c * SS]; sum += v * v; }
    float inv = 1.0f / fmaxf(sqrtf(sum), 1e-12f);
    __bf16* q = dst + (size_t)bh * DK * SS + s;
#pragma unroll 8
    for (int c = 0; c < DK; ++c) q[(size_t)c * SS] = f2bf(p[(size_t)c * SS] * inv);
}

// ---------------------------------------------------------------------------
// 4) streaming sim + top-9:  one wave per (bh, 16-row block of i)
//    sim[i,j] = sum_d kn[d,i]*qn[d,j], relu, running top-9 per row over j.
//    kn-block staged once (A frags hoisted); qn tiles streamed through LDS.
//    sim (512 MB if materialized) never touches HBM.
// ---------------------------------------------------------------------------
#define LDST 72   // 144B rows: 16B aligned

__global__ __launch_bounds__(32) void topk_kernel(
    const __bf16* __restrict__ kn, const __bf16* __restrict__ qn,
    int* __restrict__ idx) {
    __shared__ __align__(16) __bf16 Al[16 * LDST];
    __shared__ __align__(16) __bf16 Bl[16 * LDST];
    __shared__ float stile[16][17];

    const int lane = threadIdx.x;
    const int bh = blockIdx.y;
    const int i0 = blockIdx.x * 16;
    const __bf16* kb = kn + (size_t)bh * DK * SS;
    const __bf16* qb = qn + (size_t)bh * DK * SS;

    for (int e = lane; e < 16 * DK; e += 32) {
        int d = e >> 4, m = e & 15;
        Al[m * LDST + d] = kb[(size_t)d * SS + i0 + m];
    }
    __syncthreads();
    v16bf a0 = frag_a<LDST>(Al, 0, 0, lane);   // d = 0..31
    v16bf a1 = frag_a<LDST>(Al, 0, 1, lane);   // d = 32..63

    float tv[KK_TOP];
    int   ti[KK_TOP];
    for (int p = 0; p < KK_TOP; ++p) { tv[p] = -1.0f; ti[p] = 0; }

    for (int jt = 0; jt < SS / 16; ++jt) {
        int j0 = jt * 16;
        for (int e = lane; e < 16 * DK; e += 32) {
            int d = e >> 4, n = e & 15;
            Bl[n * LDST + d] = qb[(size_t)d * SS + j0 + n];
        }
        __syncthreads();

        v16bf b0 = frag_b<LDST>(Bl, 0, 0, lane);
        v16bf b1 = frag_b<LDST>(Bl, 0, 1, lane);
        v8f c = {};
        c = wmma_bf16(a0, b0, c);
        c = wmma_bf16(a1, b1, c);

#pragma unroll
        for (int r = 0; r < 8; ++r) {
            float v = c[r] > 0.f ? c[r] : 0.f;               // relu clamp
            int row = r + ((lane < 16) ? 0 : 8);
            stile[row][lane & 15] = v;
        }
        __syncthreads();

        if (lane < 16) {
            for (int cc = 0; cc < 16; ++cc) {
                float v = stile[lane][cc];
                if (v > tv[KK_TOP - 1]) {                    // insertion top-9
                    int p = KK_TOP - 1;
                    while (p > 0 && v > tv[p - 1]) {
                        tv[p] = tv[p - 1]; ti[p] = ti[p - 1]; --p;
                    }
                    tv[p] = v; ti[p] = j0 + cc;
                }
            }
        }
        __syncthreads();
    }

    if (lane < 16) {
        int* o = idx + (size_t)(bh * SS + i0 + lane) * KK_TOP;
        for (int p = 0; p < KK_TOP; ++p) o[p] = ti[p];
    }
}

// ---------------------------------------------------------------------------
// 5) gather neighbors + grouped conv (kernel=9, stride=9):
//    out[bh][o][i] = cb[o] + sum_{kk,c} cw[o][c][kk] * v[bh][c][ idx[bh][i][kk] ]
//    block = 256 threads: 4 output columns x 64 output channels.
//    The v-column gather is a pure memory->LDS scatter/gather: routed through
//    the async engine (no VGPR transit).
// ---------------------------------------------------------------------------
__global__ __launch_bounds__(256) void gather_conv_kernel(
    const float* __restrict__ v, const int* __restrict__ idx,
    const float* __restrict__ cw, const float* __restrict__ cb,
    float* __restrict__ out) {
    __shared__ float vg[4][KK_TOP][DK];
    const int bh = blockIdx.y;
    const int i0 = blockIdx.x * 4;
    const int tid = threadIdx.x;

    for (int e = tid; e < 4 * KK_TOP * DK; e += 256) {
        int ii  = e / (KK_TOP * DK);
        int rem = e - ii * (KK_TOP * DK);
        int kk  = rem >> 6;
        int c   = rem & (DK - 1);
        int j   = idx[(size_t)(bh * SS + i0 + ii) * KK_TOP + kk];
        const float* gp = &v[(size_t)(bh * DK + c) * SS + j];
#if USE_ASYNC_LDS
        __builtin_amdgcn_global_load_async_to_lds_b32(
            (AS1 int*)gp, (AS3 int*)&vg[ii][kk][c], 0, 0);
#else
        vg[ii][kk][c] = *gp;
#endif
    }
    wait_async_all();
    __syncthreads();

    int o  = tid & 63;
    int ii = tid >> 6;
    float acc = cb[o];
    for (int kk = 0; kk < KK_TOP; ++kk) {
#pragma unroll 8
        for (int c = 0; c < DK; ++c)
            acc += cw[(size_t)(o * DK + c) * KK_TOP + kk] * vg[ii][kk][c];
    }
    out[(size_t)(bh * DK + o) * SS + i0 + ii] = acc;
}

// ---------------------------------------------------------------------------
// 6) repack: faithful torch view-scramble -> y bf16 [MM][DD]
//    y[b][s2][h*64+d] = conv_out[b*8+h][f&63][f>>6],  f = d*2048 + s2
// ---------------------------------------------------------------------------
__global__ void repack_kernel(const float* __restrict__ co,
                              __bf16* __restrict__ y) {
    int t = blockIdx.x * blockDim.x + threadIdx.x;    // < MM*DD = 4M
    int b   = t >> 20;
    int rem = t & ((1 << 20) - 1);
    int s2  = rem >> 9;
    int col = rem & (DD - 1);
    int h = col >> 6, d = col & (DK - 1);
    int f = (d << 11) + s2;
    int c2 = f & (DK - 1), i2 = f >> 6;
    y[t] = f2bf(co[(size_t)(((b << 3) + h) * DK + c2) * SS + i2]);
}

// ---------------------------------------------------------------------------
// launcher
// ---------------------------------------------------------------------------
extern "C" void kernel_launch(void* const* d_in, const int* in_sizes, int n_in,
                              void* d_out, int out_size, void* d_ws, size_t ws_size,
                              hipStream_t stream) {
    const float* x      = (const float*)d_in[0];
    const float* Wq_w   = (const float*)d_in[1];
    const float* Wq_b   = (const float*)d_in[2];
    const float* Wk_w   = (const float*)d_in[3];
    const float* Wk_b   = (const float*)d_in[4];
    const float* Wv_w   = (const float*)d_in[5];
    const float* Wv_b   = (const float*)d_in[6];
    const float* Wo_w   = (const float*)d_in[7];
    const float* Wo_b   = (const float*)d_in[8];
    const float* conv_w = (const float*)d_in[9];
    const float* conv_b = (const float*)d_in[10];
    float* out = (float*)d_out;

    // workspace carve-up (256B aligned)
    char* ws = (char*)d_ws;
    size_t off = 0;
    auto take = [&](size_t bytes) {
        char* p = ws + off;
        off += (bytes + 255) & ~(size_t)255;
        return p;
    };
    __bf16* xb  = (__bf16*)take((size_t)MM * DD * 2);          // 8 MB
    __bf16* wqb = (__bf16*)take((size_t)DD * DD * 2);          // 512 KB
    __bf16* wkb = (__bf16*)take((size_t)DD * DD * 2);
    __bf16* wvb = (__bf16*)take((size_t)DD * DD * 2);
    __bf16* wob = (__bf16*)take((size_t)DD * DD * 2);
    float*  q   = (float*) take((size_t)BH * DK * SS * 4);     // 16 MB
    float*  k   = (float*) take((size_t)BH * DK * SS * 4);
    float*  v   = (float*) take((size_t)BH * DK * SS * 4);
    __bf16* qn  = (__bf16*)take((size_t)BH * DK * SS * 2);     // 8 MB
    __bf16* kn  = (__bf16*)take((size_t)BH * DK * SS * 2);
    int*    idx = (int*)   take((size_t)BH * SS * KK_TOP * 4);
    float*  co  = (float*) take((size_t)BH * DK * SS * 4);     // 16 MB
    __bf16* ybf = (__bf16*)take((size_t)MM * DD * 2);          // 8 MB
    (void)ws_size; (void)in_sizes; (void)n_in; (void)out_size;

    // 1) convert activations + weights to bf16
    {
        int n = MM * DD;
        cvt_bf16_kernel<<<(n + 255) / 256, 256, 0, stream>>>(x, xb, n);
        n = DD * DD;
        cvt_bf16_kernel<<<(n + 255) / 256, 256, 0, stream>>>(Wq_w, wqb, n);
        cvt_bf16_kernel<<<(n + 255) / 256, 256, 0, stream>>>(Wk_w, wkb, n);
        cvt_bf16_kernel<<<(n + 255) / 256, 256, 0, stream>>>(Wv_w, wvb, n);
        cvt_bf16_kernel<<<(n + 255) / 256, 256, 0, stream>>>(Wo_w, wob, n);
    }

    // 2) QKV projections (WMMA + async double-buffered staging),
    //    written directly in [BH][DK][SS] layout
    {
        dim3 grid(DD / 64, MM / 64);
        gemm_bf16_kernel<<<grid, 128, 0, stream>>>(xb, wqb, Wq_b, q, 1);
        gemm_bf16_kernel<<<grid, 128, 0, stream>>>(xb, wkb, Wk_b, k, 1);
        gemm_bf16_kernel<<<grid, 128, 0, stream>>>(xb, wvb, Wv_b, v, 1);
    }

    // 3) cosine-normalize q,k over channel dim -> bf16
    {
        dim3 grid(SS / 256, BH);
        normalize_kernel<<<grid, 256, 0, stream>>>(k, kn);
        normalize_kernel<<<grid, 256, 0, stream>>>(q, qn);
    }

    // 4) streaming sim (WMMA) + top-9 indices
    {
        dim3 grid(SS / 16, BH);
        topk_kernel<<<grid, 32, 0, stream>>>(kn, qn, idx);
    }

    // 5) gather + grouped conv (async gather into LDS)
    {
        dim3 grid(SS / 4, BH);
        gather_conv_kernel<<<grid, 256, 0, stream>>>(v, idx, conv_w, conv_b, co);
    }

    // 6) repack with the reference's view-scramble, convert to bf16
    {
        int n = MM * DD;
        repack_kernel<<<(n + 255) / 256, 256, 0, stream>>>(co, ybf);
    }

    // 7) output projection (WMMA) -> d_out fp32
    {
        dim3 grid(DD / 64, MM / 64);
        gemm_bf16_kernel<<<grid, 128, 0, stream>>>(ybf, wob, Wo_b, out, 0);
    }
}
